// PointNet2_50732153700750
// MI455X (gfx1250) — matrix operand order repaired and checked by reading, not compiled, not run on hardware
//
#include <hip/hip_runtime.h>
#include <hip/hip_bf16.h>
#include <math.h>

// ---------------------------------------------------------------------------
// PointNet++ part-seg forward for MI455X (gfx1250, wave32, WMMA).
// Dense math: padded-bf16 staging + v_wmma_f32_16x16x32_bf16, 16x64 wave tiles.
// ---------------------------------------------------------------------------

typedef __attribute__((ext_vector_type(16))) __bf16 v16bf;
typedef __attribute__((ext_vector_type(8)))  __bf16 v8bf;
typedef __attribute__((ext_vector_type(8)))  float  v8f;

static constexpr int   BB    = 8;
static constexpr int   NN    = 8192;
static constexpr int   JJ    = 21;
static constexpr int   C_L0  = 3 + 4 * JJ;   // 87
static constexpr float KERN  = 0.8f;
static constexpr float BNEPS = 1e-5f;

// ---------------------------------------------------------------------------
// Cast fp32 [R,C] -> zero-padded bf16 [Rp,Cp].
// ---------------------------------------------------------------------------
__global__ void k_cast_pad(const float* __restrict__ X, __bf16* __restrict__ Xb,
                           int R, int C, int Rp, int Cp)
{
    const size_t i = (size_t)blockIdx.x * blockDim.x + threadIdx.x;
    if (i >= (size_t)Rp * Cp) return;
    const int r = (int)(i / Cp), c = (int)(i % Cp);
    const float v = (r < R && c < C) ? X[(size_t)r * C + c] : 0.0f;
    Xb[i] = (__bf16)v;
}

// ---------------------------------------------------------------------------
// GEMM: Y[M,Co] = Xb[M,Cip](bf16) * Wb[Np,Cip](bf16)^T + bias[Co]
// One wave32 computes a 16x64 tile: 4 accumulators, A fragment reused 4x.
// Fragment <-> memory mapping (this code's contiguous k order, 16-bit operands):
//   A (16x32): lane l: M = l&15; elems 0..15 = K 16*(l>>4)+0..15    (1x b256)
//   B (32x16): lane l: N = l&15; elems 0..15 = K 16*(l>>4)+0..15    (2x b128)
//   C/D f32  : VGPR r -> M = r + 8*(l>>4), N = l&15
// M multiple of 16, Cip multiple of 32, Wb padded to Np=64*tilesN64 rows.
// ---------------------------------------------------------------------------
__global__ void k_gemm_wmma(const __bf16* __restrict__ Xb,
                            const __bf16* __restrict__ Wb,
                            const float* __restrict__ bias,
                            float* __restrict__ Y,
                            int M, int Cip, int Co, int tilesN64)
{
    const int tilesM = M >> 4;
    const int wave = blockIdx.x * (blockDim.x >> 5) + (threadIdx.x >> 5);
    if (wave >= tilesM * tilesN64) return;            // wave-uniform exit

    const int tm = wave / tilesN64;
    const int tn = wave % tilesN64;
    const int lane = threadIdx.x & 31;
    const int lh   = lane >> 4;
    const int l15  = lane & 15;
    const int mg    = tm * 16 + l15;
    const int nbase = tn * 64;

    v8f acc[4];
#pragma unroll
    for (int t = 0; t < 4; ++t) {
        const int ng = nbase + t * 16 + l15;
        const float bv = (ng < Co) ? bias[ng] : 0.0f;
#pragma unroll
        for (int r = 0; r < 8; ++r) acc[t][r] = bv;
    }

    const __bf16* pa  = Xb + (size_t)mg * Cip + 16 * lh;
    const __bf16* pb0 = Wb + (size_t)(nbase + l15) * Cip + 16 * lh;
    const size_t  bstride = (size_t)16 * Cip;         // 16 rows of W per sub-tile

    for (int k0 = 0; k0 < Cip; k0 += 32) {
        const v16bf a = *(const v16bf*)(pa + k0);
#pragma unroll
        for (int t = 0; t < 4; ++t) {
            const v16bf b = *(const v16bf*)(pb0 + (size_t)t * bstride + k0);
            acc[t] = __builtin_amdgcn_wmma_f32_16x16x32_bf16(false, a, false, b,
                                                             (short)0, acc[t],
                                                             false, false);
        }
    }

#pragma unroll
    for (int t = 0; t < 4; ++t) {
        const int ng = nbase + t * 16 + l15;
        if (ng < Co) {
#pragma unroll
            for (int r = 0; r < 8; ++r) {
                const int m = tm * 16 + r + lh * 8;   // M always multiple of 16
                Y[(size_t)m * Co + ng] = acc[t][r];
            }
        }
    }
}

// ---------------------------------------------------------------------------
// Training-mode BN helpers (mean/var over all rows, per channel).
// ---------------------------------------------------------------------------
__global__ void k_bn_stats(const float* __restrict__ Y, float* __restrict__ mean,
                           float* __restrict__ var, int M, int Co)
{
    const int c = blockIdx.x;
    __shared__ float ssum[256];
    __shared__ float ssq[256];
    float s = 0.f, q = 0.f;
    for (int m = threadIdx.x; m < M; m += blockDim.x) {
        const float v = Y[(size_t)m * Co + c];
        s += v; q += v * v;
    }
    ssum[threadIdx.x] = s; ssq[threadIdx.x] = q;
    __syncthreads();
    for (int off = 128; off > 0; off >>= 1) {
        if (threadIdx.x < off) {
            ssum[threadIdx.x] += ssum[threadIdx.x + off];
            ssq[threadIdx.x]  += ssq[threadIdx.x + off];
        }
        __syncthreads();
    }
    if (threadIdx.x == 0) {
        const float m_ = ssum[0] / (float)M;
        float v_ = ssq[0] / (float)M - m_ * m_;
        mean[c] = m_;
        var[c]  = fmaxf(v_, 0.f);
    }
}

__global__ void k_bn_relu(float* __restrict__ Y, const float* __restrict__ mean,
                          const float* __restrict__ var, const float* __restrict__ g,
                          const float* __restrict__ beta, int M, int Co)
{
    const size_t i = (size_t)blockIdx.x * blockDim.x + threadIdx.x;
    if (i >= (size_t)M * Co) return;
    const int c = (int)(i % Co);
    const float x = (Y[i] - mean[c]) * rsqrtf(var[c] + BNEPS) * g[c] + beta[c];
    Y[i] = fmaxf(x, 0.f);
}

// ---------------------------------------------------------------------------
// joint2pcloffset + concat(xyz, feat) -> l0_points [B,N,87]
// ---------------------------------------------------------------------------
__global__ void k_l0(const float* __restrict__ xyz, const float* __restrict__ joint,
                     float* __restrict__ l0)
{
    const int i = blockIdx.x * blockDim.x + threadIdx.x;
    if (i >= BB * NN) return;
    const int b = i / NN;
    const float px = xyz[(size_t)i * 3 + 0];
    const float py = xyz[(size_t)i * 3 + 1];
    const float pz = xyz[(size_t)i * 3 + 2];
    float* o = l0 + (size_t)i * C_L0;
    o[0] = px; o[1] = py; o[2] = pz;
#pragma unroll 1
    for (int j = 0; j < JJ; ++j) {
        const float ox = joint[((size_t)b * JJ + j) * 3 + 0] - px;
        const float oy = joint[((size_t)b * JJ + j) * 3 + 1] - py;
        const float oz = joint[((size_t)b * JJ + j) * 3 + 2] - pz;
        const float dis = sqrtf(ox * ox + oy * oy + oz * oz);
        const float inv = 1.0f / dis;
        float d = (KERN - dis) / KERN;
        const float mask = (d >= 0.f) ? 1.f : 0.f;
        d *= mask;
        o[3 + j * 3 + 0] = ox * inv * mask;
        o[3 + j * 3 + 1] = oy * inv * mask;
        o[3 + j * 3 + 2] = oz * inv * mask;
        o[3 + 3 * JJ + j] = d;
    }
}

// ---------------------------------------------------------------------------
// Farthest point sampling: one block (256 thr) per batch, register dist cache.
// ---------------------------------------------------------------------------
__global__ void k_fps(const float* __restrict__ xyz, int* __restrict__ fidx,
                      int Nsrc, int npoint)
{
    const int b = blockIdx.x;
    const float* P = xyz + (size_t)b * Nsrc * 3;
    __shared__ float cent[3];
    __shared__ float smax[256];
    __shared__ int   sarg[256];
    const int PT = Nsrc / 256;           // 32 (N=8192) or 2 (N=512)
    float dist[32];
#pragma unroll
    for (int j = 0; j < 32; ++j) dist[j] = 1e10f;
    int far = 0;
    for (int it = 0; it < npoint; ++it) {
        if (threadIdx.x == 0) {
            fidx[b * npoint + it] = far;
            cent[0] = P[(size_t)far * 3 + 0];
            cent[1] = P[(size_t)far * 3 + 1];
            cent[2] = P[(size_t)far * 3 + 2];
        }
        __syncthreads();
        const float cx = cent[0], cy = cent[1], cz = cent[2];
        float bm = -1.f; int ba = 0;
        for (int j = 0; j < PT; ++j) {
            const int n = threadIdx.x * PT + j;
            const float dx = P[(size_t)n * 3 + 0] - cx;
            const float dy = P[(size_t)n * 3 + 1] - cy;
            const float dz = P[(size_t)n * 3 + 2] - cz;
            const float d = dx * dx + dy * dy + dz * dz;
            dist[j] = fminf(dist[j], d);
            if (dist[j] > bm) { bm = dist[j]; ba = n; }
        }
        smax[threadIdx.x] = bm; sarg[threadIdx.x] = ba;
        __syncthreads();
        for (int off = 128; off > 0; off >>= 1) {
            if (threadIdx.x < off) {
                if (smax[threadIdx.x + off] > smax[threadIdx.x]) {
                    smax[threadIdx.x] = smax[threadIdx.x + off];
                    sarg[threadIdx.x] = sarg[threadIdx.x + off];
                }
            }
            __syncthreads();
        }
        far = sarg[0];
        __syncthreads();
    }
}

__global__ void k_gather3(const float* __restrict__ xyz, const int* __restrict__ idx,
                          float* __restrict__ out, int S, int Nsrc)
{
    const int i = blockIdx.x * blockDim.x + threadIdx.x;
    if (i >= BB * S) return;
    const int b = i / S;
    const int n = idx[i];
    const float* p = xyz + ((size_t)b * Nsrc + n) * 3;
    out[(size_t)i * 3 + 0] = p[0];
    out[(size_t)i * 3 + 1] = p[1];
    out[(size_t)i * 3 + 2] = p[2];
}

// Ball query: first K in-index-order neighbors within radius; pad with first.
__global__ void k_qball(const float* __restrict__ xyz, const float* __restrict__ nxyz,
                        int* __restrict__ gidx, int S, int K, float r2, int Nsrc)
{
    const int i = blockIdx.x * blockDim.x + threadIdx.x;
    if (i >= BB * S) return;
    const int b = i / S;
    const float* P = xyz + (size_t)b * Nsrc * 3;
    const float cx = nxyz[(size_t)i * 3 + 0];
    const float cy = nxyz[(size_t)i * 3 + 1];
    const float cz = nxyz[(size_t)i * 3 + 2];
    int cnt = 0, first = -1;
    for (int n = 0; n < Nsrc && cnt < K; ++n) {
        const float dx = P[(size_t)n * 3 + 0] - cx;
        const float dy = P[(size_t)n * 3 + 1] - cy;
        const float dz = P[(size_t)n * 3 + 2] - cz;
        if (dx * dx + dy * dy + dz * dz <= r2) {
            if (first < 0) first = n;
            gidx[(size_t)i * K + cnt++] = n;
        }
    }
    if (first < 0) first = 0;
    for (; cnt < K; ++cnt) gidx[(size_t)i * K + cnt] = first;
}

// Grouping: out[b,s,k,:] = [xyz[gi]-center, points[gi]]
__global__ void k_group(const float* __restrict__ xyz, const float* __restrict__ points,
                        const float* __restrict__ nxyz, const int* __restrict__ gidx,
                        float* __restrict__ out, int S, int K, int Cin, int Nsrc)
{
    const int i = blockIdx.x * blockDim.x + threadIdx.x;
    if (i >= BB * S * K) return;
    const int bs = i / K;
    const int b  = bs / S;
    const int n  = gidx[i];
    const float* p = xyz + ((size_t)b * Nsrc + n) * 3;
    const float* c = nxyz + (size_t)bs * 3;
    float* o = out + (size_t)i * (3 + Cin);
    o[0] = p[0] - c[0]; o[1] = p[1] - c[1]; o[2] = p[2] - c[2];
    const float* f = points + ((size_t)b * Nsrc + n) * Cin;
    for (int ch = 0; ch < Cin; ++ch) o[3 + ch] = f[ch];
}

// Row-wise concat of two channel blocks.
__global__ void k_concat2(const float* __restrict__ A, int Ca,
                          const float* __restrict__ Bp, int Cb,
                          float* __restrict__ out, int rows)
{
    const int i = blockIdx.x * blockDim.x + threadIdx.x;
    if (i >= rows) return;
    float* o = out + (size_t)i * (Ca + Cb);
    const float* a = A + (size_t)i * Ca;
    const float* bq = Bp + (size_t)i * Cb;
    for (int c = 0; c < Ca; ++c) o[c] = a[c];
    for (int c = 0; c < Cb; ++c) o[Ca + c] = bq[c];
}

// Max over the K (neighbor) axis: X[S,K,C] -> Y[S,C]
__global__ void k_maxpool(const float* __restrict__ X, float* __restrict__ Y,
                          int S, int K, int C)
{
    const size_t i = (size_t)blockIdx.x * blockDim.x + threadIdx.x;
    if (i >= (size_t)S * C) return;
    const int s = (int)(i / C), c = (int)(i % C);
    float m = -3.0e38f;
    for (int k = 0; k < K; ++k) m = fmaxf(m, X[((size_t)s * K + k) * C + c]);
    Y[i] = m;
}

// Feature propagation input: out[b,n,:] = [points1, 3NN-interp(points2)]
__global__ void k_fp(const float* __restrict__ xyz1, const float* __restrict__ xyz2,
                     const float* __restrict__ p1, const float* __restrict__ p2,
                     float* __restrict__ out, int N1, int S2, int C1, int C2)
{
    const int i = blockIdx.x * blockDim.x + threadIdx.x;
    if (i >= BB * N1) return;
    const int b = i / N1;
    float* o = out + (size_t)i * (C1 + C2);
    const float* f1 = p1 + (size_t)i * C1;
    for (int c = 0; c < C1; ++c) o[c] = f1[c];
    if (S2 == 1) {
        const float* f2 = p2 + (size_t)b * C2;
        for (int c = 0; c < C2; ++c) o[C1 + c] = f2[c];
    } else {
        const float cx = xyz1[(size_t)i * 3 + 0];
        const float cy = xyz1[(size_t)i * 3 + 1];
        const float cz = xyz1[(size_t)i * 3 + 2];
        const float* Q = xyz2 + (size_t)b * S2 * 3;
        float d0 = 1e30f, d1 = 1e30f, d2 = 1e30f;
        int i0 = 0, i1 = 0, i2 = 0;
        for (int s = 0; s < S2; ++s) {
            const float dx = Q[(size_t)s * 3 + 0] - cx;
            const float dy = Q[(size_t)s * 3 + 1] - cy;
            const float dz = Q[(size_t)s * 3 + 2] - cz;
            const float d = dx * dx + dy * dy + dz * dz;
            if (d < d0)      { d2 = d1; i2 = i1; d1 = d0; i1 = i0; d0 = d; i0 = s; }
            else if (d < d1) { d2 = d1; i2 = i1; d1 = d;  i1 = s; }
            else if (d < d2) { d2 = d;  i2 = s; }
        }
        float w0 = 1.f / (d0 + 1e-8f), w1 = 1.f / (d1 + 1e-8f), w2 = 1.f / (d2 + 1e-8f);
        const float ws = w0 + w1 + w2;
        w0 /= ws; w1 /= ws; w2 /= ws;
        const float* q0 = p2 + ((size_t)b * S2 + i0) * C2;
        const float* q1 = p2 + ((size_t)b * S2 + i1) * C2;
        const float* q2 = p2 + ((size_t)b * S2 + i2) * C2;
        for (int c = 0; c < C2; ++c)
            o[C1 + c] = w0 * q0[c] + w1 * q1[c] + w2 * q2[c];
    }
}

// ---------------------------------------------------------------------------
// Host orchestration
// ---------------------------------------------------------------------------
struct LayerP { const float *w, *b, *g, *e; int ci, co; };

extern "C" void kernel_launch(void* const* d_in, const int* in_sizes, int n_in,
                              void* d_out, int out_size, void* d_ws, size_t ws_size,
                              hipStream_t stream)
{
    (void)in_sizes; (void)n_in; (void)out_size; (void)ws_size;
    const float* xyz   = (const float*)d_in[0];
    const float* joint = (const float*)d_in[1];

    int pi = 2;
    auto getL = [&](int ci, int co) -> LayerP {
        LayerP L;
        L.w = (const float*)d_in[pi++]; L.b = (const float*)d_in[pi++];
        L.g = (const float*)d_in[pi++]; L.e = (const float*)d_in[pi++];
        L.ci = ci; L.co = co; return L;
    };
    LayerP sa1L[3] = { getL(90, 64),   getL(64, 64),   getL(64, 128)  };
    LayerP sa2L[3] = { getL(131, 128), getL(128, 128), getL(128, 256) };
    LayerP sa3L[3] = { getL(259, 256), getL(256, 512), getL(512, 1024)};
    LayerP fp3L[2] = { getL(1280, 256), getL(256, 256) };
    LayerP fp2L[2] = { getL(384, 256),  getL(256, 128) };
    LayerP fp1L[3] = { getL(218, 128),  getL(128, 128), getL(128, 128) };
    LayerP headL   = getL(128, 128);
    const float* out_w = (const float*)d_in[pi++];
    const float* out_b = (const float*)d_in[pi++];

    // ---- workspace carve (256B-aligned byte allocator) ----
    char* wsb = (char*)d_ws;
    size_t cur = 0;
    auto allocB = [&](size_t bytes) -> void* {
        void* p = wsb + cur;
        cur += (bytes + 255) & ~(size_t)255;
        return p;
    };

    float*  l0     = (float*)allocB((size_t)BB * NN * C_L0 * 4);
    float*  mean   = (float*)allocB(1024 * 4);
    float*  var    = (float*)allocB(1024 * 4);
    float*  bufA   = (float*)allocB((size_t)16777216 * 4);
    float*  bufB   = (float*)allocB((size_t)16777216 * 4);
    float*  Gbuf   = (float*)allocB((size_t)BB * NN * 218 * 4);
    float*  catXL0 = (float*)allocB((size_t)BB * NN * 90 * 4);
    float*  l1x    = (float*)allocB((size_t)BB * 512 * 3 * 4);
    float*  l1p    = (float*)allocB((size_t)BB * 512 * 128 * 4);
    float*  l2x    = (float*)allocB((size_t)BB * 128 * 3 * 4);
    float*  l2p    = (float*)allocB((size_t)BB * 128 * 256 * 4);
    float*  l3p    = (float*)allocB((size_t)BB * 1024 * 4);
    int*    fidx   = (int*)allocB((size_t)BB * 512 * 4);
    int*    gidx   = (int*)allocB((size_t)BB * 512 * 32 * 4);
    __bf16* Xb     = (__bf16*)allocB((size_t)16777216 * 2);   // activations, bf16 padded
    __bf16* Wb     = (__bf16*)allocB((size_t)1572864 * 2);    // weights, bf16 padded

    auto runLayer = [&](const float* X, float* Y, int M, const LayerP& L, bool bn) {
        const int Cip = ((L.ci + 31) / 32) * 32;
        const int tilesN64 = (L.co + 63) / 64;
        const int Np = tilesN64 * 64;
        const size_t xel = (size_t)M * Cip;
        const size_t wel = (size_t)Np * Cip;
        k_cast_pad<<<(int)((xel + 255) / 256), 256, 0, stream>>>(X, Xb, M, L.ci, M, Cip);
        k_cast_pad<<<(int)((wel + 255) / 256), 256, 0, stream>>>(L.w, Wb, L.co, L.ci, Np, Cip);
        const int tiles = (M / 16) * tilesN64;
        k_gemm_wmma<<<(tiles + 3) / 4, 128, 0, stream>>>(Xb, Wb, L.b, Y, M, Cip, L.co, tilesN64);
        if (bn) {
            k_bn_stats<<<L.co, 256, 0, stream>>>(Y, mean, var, M, L.co);
            const size_t tot = (size_t)M * L.co;
            k_bn_relu<<<(int)((tot + 255) / 256), 256, 0, stream>>>(Y, mean, var, L.g, L.e, M, L.co);
        }
    };

    // ---- stage 0: l0_points = [xyz, joint2pcloffset] ----
    k_l0<<<(BB * NN + 255) / 256, 256, 0, stream>>>(xyz, joint, l0);

    // ---- SA1: N=8192 -> S=512, K=32, r=0.2 ----
    k_fps<<<BB, 256, 0, stream>>>(xyz, fidx, NN, 512);
    k_gather3<<<(BB * 512 + 255) / 256, 256, 0, stream>>>(xyz, fidx, l1x, 512, NN);
    k_qball<<<(BB * 512 + 255) / 256, 256, 0, stream>>>(xyz, l1x, gidx, 512, 32, 0.04f, NN);
    {
        const int M = BB * 512 * 32;
        k_group<<<(M + 255) / 256, 256, 0, stream>>>(xyz, l0, l1x, gidx, Gbuf, 512, 32, C_L0, NN);
        runLayer(Gbuf, bufA, M, sa1L[0], true);
        runLayer(bufA, bufB, M, sa1L[1], true);
        runLayer(bufB, bufA, M, sa1L[2], true);
        k_maxpool<<<(int)(((size_t)BB * 512 * 128 + 255) / 256), 256, 0, stream>>>(bufA, l1p, BB * 512, 32, 128);
    }

    // ---- SA2: S=512 -> 128, K=64, r=0.4 ----
    k_fps<<<BB, 256, 0, stream>>>(l1x, fidx, 512, 128);
    k_gather3<<<(BB * 128 + 255) / 256, 256, 0, stream>>>(l1x, fidx, l2x, 128, 512);
    k_qball<<<(BB * 128 + 255) / 256, 256, 0, stream>>>(l1x, l2x, gidx, 128, 64, 0.16f, 512);
    {
        const int M = BB * 128 * 64;
        k_group<<<(M + 255) / 256, 256, 0, stream>>>(l1x, l1p, l2x, gidx, Gbuf, 128, 64, 128, 512);
        runLayer(Gbuf, bufA, M, sa2L[0], true);
        runLayer(bufA, bufB, M, sa2L[1], true);
        runLayer(bufB, bufA, M, sa2L[2], true);
        k_maxpool<<<(int)(((size_t)BB * 128 * 256 + 255) / 256), 256, 0, stream>>>(bufA, l2p, BB * 128, 64, 256);
    }

    // ---- SA3: group_all over the 128 points ----
    {
        const int M = BB * 128;
        k_concat2<<<(M + 255) / 256, 256, 0, stream>>>(l2x, 3, l2p, 256, Gbuf, M);
        runLayer(Gbuf, bufA, M, sa3L[0], true);
        runLayer(bufA, bufB, M, sa3L[1], true);
        runLayer(bufB, bufA, M, sa3L[2], true);
        k_maxpool<<<(BB * 1024 + 255) / 256, 256, 0, stream>>>(bufA, l3p, BB, 128, 1024);
    }

    // ---- FP3: broadcast l3 -> 128 pts, concat with l2p, MLP [1280->256->256] ----
    {
        const int M = BB * 128;
        k_fp<<<(M + 255) / 256, 256, 0, stream>>>(l2x, l2x, l2p, l3p, Gbuf, 128, 1, 256, 1024);
        runLayer(Gbuf, bufA, M, fp3L[0], true);
        runLayer(bufA, bufB, M, fp3L[1], true);          // new l2 feats in bufB
    }

    // ---- FP2: 3-NN interp 128 -> 512, concat with l1p, MLP [384->256->128] ----
    {
        const int M = BB * 512;
        k_fp<<<(M + 255) / 256, 256, 0, stream>>>(l1x, l2x, l1p, bufB, Gbuf, 512, 128, 128, 256);
        runLayer(Gbuf, bufA, M, fp2L[0], true);
        runLayer(bufA, bufB, M, fp2L[1], true);          // new l1 feats in bufB
    }

    // ---- FP1: 3-NN interp 512 -> 8192, points1 = [xyz, l0], MLP [218->128x3] ----
    {
        const int M = BB * NN;
        k_concat2<<<(M + 255) / 256, 256, 0, stream>>>(xyz, 3, l0, C_L0, catXL0, M);
        k_fp<<<(M + 255) / 256, 256, 0, stream>>>(xyz, l1x, catXL0, bufB, Gbuf, NN, 512, 90, 128);
        runLayer(Gbuf, bufA, M, fp1L[0], true);
        runLayer(bufA, bufB, M, fp1L[1], true);
        runLayer(bufB, bufA, M, fp1L[2], true);
    }

    // ---- head conv1+bn1+relu, then classifier to d_out ----
    {
        const int M = BB * NN;
        runLayer(bufA, bufB, M, headL, true);
        LayerP outL; outL.w = out_w; outL.b = out_b; outL.g = nullptr; outL.e = nullptr;
        outL.ci = 128; outL.co = JJ;                     // 21 classes
        runLayer(bufB, (float*)d_out, M, outL, false);
    }
}
